// UniSkel_59880434041306
// MI455X (gfx1250) — matrix operand order repaired and checked by reading, not compiled, not run on hardware
//
#include <hip/hip_runtime.h>

typedef __attribute__((ext_vector_type(16))) __bf16 v16bf;
typedef __attribute__((ext_vector_type(8)))  float  v8f;
typedef int v4i_gcc __attribute__((vector_size(16)));   // matches builtin's V4i
typedef unsigned short u16;
typedef unsigned int   u32;

#define BB    32
#define NJ    17
#define HW4K  4096
#define GLS   193
#define KFC   36864
#define ROWS  (BB*NJ)          // 544
#define GPAD  256              // 16 N-tiles of 16 (padded so nt 13..15 read zeros)
#define KC    128              // K-chunk staged per LDS stage
#define STAGES (KFC / KC)      // 288
#define LDSTRIDE 136           // halves per LDS row (272B: breaks bank-conflict stride)

#if __has_builtin(__builtin_amdgcn_global_load_async_to_lds_b128)
#define ASYNC_LDS 1
#endif

// ---------- bf16 helpers ----------
__device__ __forceinline__ u16 f2bf(float f) {
  u32 u = __builtin_bit_cast(u32, f);
  u32 r = u + 0x7FFFu + ((u >> 16) & 1u);      // round-to-nearest-even
  return (u16)(r >> 16);
}
__device__ __forceinline__ float bf2f(u16 h) {
  u32 u = ((u32)h) << 16;
  return __builtin_bit_cast(float, u);
}
__device__ __forceinline__ v16bf make_frag(uint4 lo, uint4 hi) {
  union { uint4 u[2]; v16bf v; } c;
  c.u[0] = lo; c.u[1] = hi;
  return c.v;
}
__device__ __forceinline__ void wait_async() {
#if defined(ASYNC_LDS)
#if __has_builtin(__builtin_amdgcn_s_wait_asynccnt)
  __builtin_amdgcn_s_wait_asynccnt(0);
#else
  asm volatile("s_wait_asynccnt 0x0" ::: "memory");
#endif
#endif
}

// ---------- prep: w2/w3 OIHW(32,32,3,3) -> bf16 [tap][n][k] ----------
__global__ void prep_wt(const float* __restrict__ w, u16* __restrict__ wt) {
  int i = blockIdx.x * blockDim.x + threadIdx.x;          // 9216
  if (i >= 9216) return;
  int tap = i / 1024, r = i % 1024, n = r >> 5, k = r & 31;
  wt[i] = f2bf(w[(n * 32 + k) * 9 + tap]);
}

// ---------- prep: fc_w (193,36864) -> bf16 padded (256,36864) ----------
__global__ void prep_fcw(const float* __restrict__ fw, u16* __restrict__ wb) {
  long long i = (long long)blockIdx.x * blockDim.x + threadIdx.x;
  if (i >= (long long)GPAD * KFC) return;
  wb[i] = (i < (long long)GLS * KFC) ? f2bf(fw[i]) : (u16)0;
}

// ---------- conv1 (3->32, 256x256, fp32 VALU, K=27) -> NHWC bf16 ----------
__global__ void conv1_relu(const float* __restrict__ x,   // (3,256,256)
                           const float* __restrict__ w1,  // (32,3,3,3)
                           const float* __restrict__ b1,
                           u16* __restrict__ out) {       // (65536,32) NHWC bf16
  __shared__ float sw[864];
  __shared__ float sb[32];
  for (int i = threadIdx.x; i < 864; i += blockDim.x) sw[i] = w1[i];
  if (threadIdx.x < 32) sb[threadIdx.x] = b1[threadIdx.x];
  __syncthreads();
  int p = blockIdx.x * blockDim.x + threadIdx.x;          // pixel 0..65535
  int y = p >> 8, xx = p & 255;
  float in[27];
  #pragma unroll
  for (int c = 0; c < 3; ++c)
    #pragma unroll
    for (int dy = 0; dy < 3; ++dy)
      #pragma unroll
      for (int dx = 0; dx < 3; ++dx) {
        int yy = y + dy - 1, xc = xx + dx - 1;
        float v = 0.f;
        if ((unsigned)yy < 256u && (unsigned)xc < 256u)
          v = x[c * 65536 + yy * 256 + xc];
        in[c * 9 + dy * 3 + dx] = v;
      }
  #pragma unroll 4
  for (int o = 0; o < 32; ++o) {
    float s = sb[o];
    #pragma unroll
    for (int k = 0; k < 27; ++k) s += sw[o * 27 + k] * in[k];
    out[(p << 5) + o] = f2bf(s > 0.f ? s : 0.f);
  }
}

// ---------- conv 32->32 3x3 implicit GEMM via WMMA bf16, +bias +ReLU ----------
// in/out: NHWC bf16 (Wd*Wd, 32). wt: [tap][n][k] bf16. One wave = 16 pixels x 32 ch.
__global__ void conv3x3_wmma(const u16* __restrict__ in,
                             const u16* __restrict__ wt,
                             const float* __restrict__ bias,
                             u16* __restrict__ out,
                             int Wd) {
  int wave = threadIdx.x >> 5;
  int lane = threadIdx.x & 31;
  int tile = blockIdx.x * (blockDim.x >> 5) + wave;
  int nTiles = (Wd * Wd) >> 4;
  if (tile >= nTiles) return;                 // wave-uniform
  int tilesPerRow = Wd >> 4;
  int ty = tile / tilesPerRow;
  int x0 = (tile % tilesPerRow) << 4;
  int m  = lane & 15;
  int hi = lane >> 4;
  int ka0 = hi ? 16 : 0;                      // A K-chunk offsets (contiguous k = 16h + s, matches B)
  int ka1 = hi ? 24 : 8;
  int nb  = lane & 15;                        // B column in tile
  int kb  = hi ? 16 : 0;                      // B K-chunk
  // wave-uniform interior test: all 9 taps for all 16 pixels stay in-bounds
  bool interior = (ty >= 1) && (ty <= Wd - 2) && (x0 >= 16) && (x0 + 16 <= Wd - 1);

  v8f acc0 = {}; v8f acc1 = {};
  #pragma unroll
  for (int tap = 0; tap < 9; ++tap) {
    int dy = tap / 3 - 1, dx = tap % 3 - 1;
    int y = ty + dy;
    int xp = x0 + m + dx;
    const u16* p = in + ((y * Wd + xp) << 5);
    uint4 alo{}, ahi{};
    if (interior) {                           // uniform branch: no EXEC churn
      alo = *(const uint4*)(p + ka0);
      ahi = *(const uint4*)(p + ka1);
    } else if ((unsigned)y < (unsigned)Wd && (unsigned)xp < (unsigned)Wd) {
      alo = *(const uint4*)(p + ka0);
      ahi = *(const uint4*)(p + ka1);
    }
    v16bf a = make_frag(alo, ahi);
    const u16* wp0 = wt + tap * 1024 + (nb << 5) + kb;
    const u16* wp1 = wt + tap * 1024 + ((nb + 16) << 5) + kb;
    v16bf b0 = make_frag(*(const uint4*)(wp0), *(const uint4*)(wp0 + 8));
    v16bf b1 = make_frag(*(const uint4*)(wp1), *(const uint4*)(wp1 + 8));
    acc0 = __builtin_amdgcn_wmma_f32_16x16x32_bf16(false, a, false, b0, (short)0, acc0, false, false);
    acc1 = __builtin_amdgcn_wmma_f32_16x16x32_bf16(false, a, false, b1, (short)0, acc1, false, false);
  }
  float bi0 = bias[nb], bi1 = bias[nb + 16];
  int pixbase = tile << 4;
  #pragma unroll
  for (int v = 0; v < 8; ++v) {
    int mm = v + 8 * hi;                      // C/D layout: lanes>=16 hold rows M+8
    float f0 = acc0[v] + bi0; if (f0 < 0.f) f0 = 0.f;
    float f1 = acc1[v] + bi1; if (f1 < 0.f) f1 = 0.f;
    out[((pixbase + mm) << 5) + nb]      = f2bf(f0);
    out[((pixbase + mm) << 5) + nb + 16] = f2bf(f1);
  }
}

// ---------- 2x2 maxpool on NHWC bf16 ----------
__global__ void pool2_bf16(const u16* __restrict__ in, u16* __restrict__ out, int Wo) {
  int idx = blockIdx.x * blockDim.x + threadIdx.x;   // p*32 + c
  int c = idx & 31, p = idx >> 5;
  if (p >= Wo * Wo) return;
  int y = p / Wo, x = p % Wo;
  int Wi = Wo * 2;
  const u16* r0 = in + ((2 * y) * Wi + 2 * x) * 32 + c;
  const u16* r1 = r0 + Wi * 32;
  float a = bf2f(r0[0]), b = bf2f(r0[32]);
  float cc = bf2f(r1[0]), d = bf2f(r1[32]);
  out[p * 32 + c] = f2bf(fmaxf(fmaxf(a, b), fmaxf(cc, d)));
}

// ---------- conv4 (32->8) + ReLU + 2x2 pool fused, fp32 out NCHW ----------
__global__ void conv4_relu_pool(const u16* __restrict__ in,   // NHWC (128*128,32) bf16
                                const float* __restrict__ w4, // (8,32,3,3) OIHW
                                const float* __restrict__ b4,
                                float* __restrict__ out) {    // (8,4096) this batch
  __shared__ float sw[2304];                  // [c][tap][k]
  for (int i = threadIdx.x; i < 2304; i += blockDim.x) {
    int c = i / 288, r = i % 288, t = r / 32, k = r % 32;
    sw[i] = w4[c * 288 + k * 9 + t];
  }
  __syncthreads();
  int idx = blockIdx.x * blockDim.x + threadIdx.x;   // c*4096 + p (32768)
  if (idx >= 8 * HW4K) return;
  int c = idx >> 12, p = idx & 4095;
  int y = p >> 6, x = p & 63;
  float best = 0.f;                                   // max of ReLU'd values >= 0
  #pragma unroll
  for (int q = 0; q < 4; ++q) {
    int yy = 2 * y + (q >> 1), xx = 2 * x + (q & 1);
    float s = b4[c];
    #pragma unroll
    for (int t = 0; t < 9; ++t) {
      int iy = yy + t / 3 - 1, ix = xx + t % 3 - 1;
      if ((unsigned)iy < 128u && (unsigned)ix < 128u) {
        const u16* pp = in + ((iy << 7) + ix) * 32;
        const float* wr = sw + (c * 9 + t) * 32;
        #pragma unroll
        for (int k = 0; k < 32; ++k) s += bf2f(pp[k]) * wr[k];
      }
    }
    best = fmaxf(best, s);
  }
  out[idx] = best;
}

// ---------- build z (544, 36864) bf16:  z[b,n, c*4096+p] ----------
__global__ void build_z(const float* __restrict__ h4,  // (32,8,4096)
                        const float* __restrict__ hm,  // (32,17,4096)
                        u16* __restrict__ z) {
  long long i = (long long)blockIdx.x * blockDim.x + threadIdx.x;
  if (i >= (long long)ROWS * KFC) return;
  int row = (int)(i / KFC);
  int k   = (int)(i % KFC);
  int b = row / NJ, n = row % NJ;
  int c = k >> 12, p = k & 4095;
  float hv = hm[(b * NJ + n) * HW4K + p];
  float v = (c == 8) ? hv : h4[(b * 8 + c) * HW4K + p] * hv;
  z[i] = f2bf(v);
}

// ---------- FC GEMM: (544 x 36864) x (36864 x 193), WMMA bf16 ----------
// Block = 4 waves. Block owns one 16-row A tile + one 64-row B panel (4 N-tiles,
// one per wave). K staged through double-buffered LDS via async-to-LDS copies
// (ASYNCcnt), overlapping stage s+1 DMA with stage s WMMA. No early returns:
// padded wb rows (nt 13..15) compute on zeros, stores are guarded.
__global__ void fc_wmma(const u16* __restrict__ z,    // (544, 36864) bf16
                        const u16* __restrict__ wb,   // (256, 36864) bf16 zero-pad
                        const float* __restrict__ fcb,
                        float* __restrict__ out) {    // (544, 193) fp32
  __shared__ u16 sA[2][16 * LDSTRIDE];
  __shared__ u16 sB[2][64 * LDSTRIDE];
  int tid  = threadIdx.x;                     // 0..127
  int wave = tid >> 5, lane = tid & 31;
  int mt  = blockIdx.x >> 2;                  // 0..33 (M tile)
  int ntg = blockIdx.x & 3;                   // 0..3  (N group of 4 tiles)
  int nt  = ntg * 4 + wave;                   // 0..15
  int hi = lane >> 4;
  int a0 = hi ? 16 : 0, a1 = hi ? 24 : 8;     // A fragment K-chunks (contiguous k = 16h + s)
  int bo = hi ? 16 : 0;                       // B fragment K-chunk

  const u16* gA = z  + (long long)(mt  * 16) * KFC;   // 16 A rows
  const u16* gB = wb + (long long)(ntg * 64) * KFC;   // 64 B rows

  // 1280 b128 transfers per stage (A:256, B:1024) / 128 threads = 10 each
  auto issue = [&](int stage, int buf) {
    int k0 = stage * KC;
    #pragma unroll
    for (int j = 0; j < 10; ++j) {
      int t = tid + j * 128;
      const u16* src;
      u16* dst;
      if (t < 256) {
        int row = t >> 4, ch = t & 15;
        src = gA + (long long)row * KFC + k0 + ch * 8;
        dst = &sA[buf][row * LDSTRIDE + ch * 8];
      } else {
        int tb = t - 256;
        int row = tb >> 4, ch = tb & 15;
        src = gB + (long long)row * KFC + k0 + ch * 8;
        dst = &sB[buf][row * LDSTRIDE + ch * 8];
      }
#if defined(ASYNC_LDS)
      __builtin_amdgcn_global_load_async_to_lds_b128(
          (__attribute__((address_space(1))) v4i_gcc*)src,
          (__attribute__((address_space(3))) v4i_gcc*)dst, 0, 0);
#else
      *(uint4*)dst = *(const uint4*)src;
#endif
    }
  };

  v8f acc = {};
  issue(0, 0);
  wait_async();
  __syncthreads();
  for (int s = 0; s < STAGES; ++s) {
    int buf = s & 1;
    if (s + 1 < STAGES) issue(s + 1, buf ^ 1);   // DMA next stage during compute
    const u16* Arow = &sA[buf][(lane & 15) * LDSTRIDE];
    const u16* Brow = &sB[buf][(wave * 16 + (lane & 15)) * LDSTRIDE];
    #pragma unroll
    for (int kk = 0; kk < KC; kk += 32) {
      uint4 alo = *(const uint4*)(Arow + kk + a0);
      uint4 ahi = *(const uint4*)(Arow + kk + a1);
      uint4 blo = *(const uint4*)(Brow + kk + bo);
      uint4 bhi = *(const uint4*)(Brow + kk + bo + 8);
      v16bf a = make_frag(alo, ahi);
      v16bf b = make_frag(blo, bhi);
      acc = __builtin_amdgcn_wmma_f32_16x16x32_bf16(false, a, false, b, (short)0, acc, false, false);
    }
    wait_async();
    __syncthreads();
  }
  int g = nt * 16 + (lane & 15);
  if (g < GLS) {
    float bias = fcb[g];
    #pragma unroll
    for (int v = 0; v < 8; ++v) {
      int row = mt * 16 + v + 8 * hi;
      out[row * GLS + g] = acc[v] + bias;
    }
  }
}

// ---------- GLES: per-batch 17x17 graph ----------
__global__ void gles_kernel(const float* __restrict__ outp,  // (32,17,193)
                            float* __restrict__ graphs) {    // (32,17,17)
  int b = blockIdx.x;
  __shared__ float F[NJ][200];
  __shared__ float Wm[NJ][NJ];
  __shared__ float drow[NJ];
  int t = threadIdx.x;                        // blockDim = 289
  for (int i = t; i < NJ * GLS; i += blockDim.x) {
    int r = i / GLS, k = i % GLS;
    F[r][k] = outp[(b * NJ + r) * GLS + k];
  }
  __syncthreads();
  int i = t / NJ, j = t % NJ;
  float dot = 0.f, si = 0.f, sj = 0.f;
  for (int k = 0; k < GLS; ++k) {
    float a = F[i][k], bb = F[j][k];
    dot += a * bb; si += a * a; sj += bb * bb;
  }
  float dij = si + sj - 2.f * dot;            // squared distance
  float w = expf(-dij);                       // exp(-D / (2*0.5))
  if (i == j) w = 0.f;
  Wm[i][j] = w;
  __syncthreads();
  for (int it = 0; it < 5; ++it) {
    if (j == 0) {
      float s = 0.f;
      for (int q = 0; q < NJ; ++q) s += Wm[i][q];
      drow[i] = s;
    }
    __syncthreads();
    float nw = Wm[i][j] / sqrtf(drow[i] * drow[j] + 1e-8f);
    __syncthreads();
    Wm[i][j] = nw;
    __syncthreads();
  }
  graphs[(b * NJ + i) * NJ + j] = 0.5f * (Wm[i][j] + Wm[j][i]);
}

// ---------- host ----------
extern "C" void kernel_launch(void* const* d_in, const int* in_sizes, int n_in,
                              void* d_out, int out_size, void* d_ws, size_t ws_size,
                              hipStream_t stream) {
  (void)in_sizes; (void)n_in; (void)out_size; (void)ws_size;
  const float* x   = (const float*)d_in[0];   // (32,3,256,256)
  const float* hm  = (const float*)d_in[1];   // (32,17,64,64)
  const float* w1  = (const float*)d_in[2];
  const float* b1  = (const float*)d_in[3];
  const float* w2  = (const float*)d_in[4];
  const float* b2  = (const float*)d_in[5];
  const float* w3  = (const float*)d_in[6];
  const float* b3  = (const float*)d_in[7];
  const float* w4  = (const float*)d_in[8];
  const float* b4  = (const float*)d_in[9];
  const float* fcw = (const float*)d_in[10];
  const float* fcb = (const float*)d_in[11];
  float* out    = (float*)d_out;                    // (544,193) fp32
  float* graphs = out + ROWS * GLS;                 // (32,17,17) fp32

  char* ws = (char*)d_ws;
  size_t off = 0;
  auto alloc = [&](size_t bytes) { size_t o = off; off = (off + bytes + 255) & ~(size_t)255; return o; };
  u16*   c1   = (u16*)(ws + alloc((size_t)65536 * 32 * 2));   // per-batch conv1 out, NHWC
  u16*   c2   = (u16*)(ws + alloc((size_t)65536 * 32 * 2));   // per-batch conv2 out
  u16*   c2p  = (u16*)(ws + alloc((size_t)16384 * 32 * 2));   // pooled
  u16*   c3   = (u16*)(ws + alloc((size_t)16384 * 32 * 2));   // conv3 out
  float* h4   = (float*)(ws + alloc((size_t)BB * 8 * HW4K * 4)); // all batches
  u16*   wt2  = (u16*)(ws + alloc((size_t)9216 * 2));
  u16*   wt3  = (u16*)(ws + alloc((size_t)9216 * 2));
  u16*   fcwb = (u16*)(ws + alloc((size_t)GPAD * KFC * 2));
  u16*   zbuf = (u16*)(ws + alloc((size_t)ROWS * KFC * 2));

  // one-time preps
  prep_wt<<<36, 256, 0, stream>>>(w2, wt2);
  prep_wt<<<36, 256, 0, stream>>>(w3, wt3);
  {
    long long n = (long long)GPAD * KFC;
    prep_fcw<<<(unsigned)((n + 255) / 256), 256, 0, stream>>>(fcw, fcwb);
  }

  // backbone, per batch (keeps 256^2 activations L2-resident)
  for (int b = 0; b < BB; ++b) {
    conv1_relu<<<256, 256, 0, stream>>>(x + (size_t)b * 3 * 65536, w1, b1, c1);
    conv3x3_wmma<<<1024, 128, 0, stream>>>(c1, wt2, b2, c2, 256);   // 4096 M-tiles
    pool2_bf16<<<2048, 256, 0, stream>>>(c2, c2p, 128);
    conv3x3_wmma<<<256, 128, 0, stream>>>(c2p, wt3, b3, c3, 128);   // 1024 M-tiles
    conv4_relu_pool<<<128, 256, 0, stream>>>(c3, w4, b4, h4 + (size_t)b * 8 * HW4K);
  }

  // z, FC (WMMA + async LDS pipeline), GLES
  {
    long long n = (long long)ROWS * KFC;
    build_z<<<(unsigned)((n + 255) / 256), 256, 0, stream>>>(h4, hm, zbuf);
  }
  fc_wmma<<<136, 128, 0, stream>>>(zbuf, fcwb, fcb, out);           // 34 x 4 blocks
  gles_kernel<<<BB, 289, 0, stream>>>(out, graphs);
}
